// FasterRCNN_42210938585718
// MI455X (gfx1250) — matrix operand order repaired and checked by reading, not compiled, not run on hardware
//
#include <hip/hip_runtime.h>

typedef __attribute__((ext_vector_type(16))) _Float16     v16h;
typedef __attribute__((ext_vector_type(8)))  float        v8f;
typedef __attribute__((ext_vector_type(8)))  unsigned int v8u;

#define HFD 50
#define WFD 50
#define NPIX 2500
#define CIN 512
#define NANCH 22500
#define NROI 128
#define NCLS 21
#define FC1K 25088
#define FCN 4096
#define BBOX_CLIP 4.135166556742356f

union Frag { v8u u; v16h h; };

static __device__ __forceinline__ float clampf(float v, float lo, float hi) {
  return fminf(fmaxf(v, lo), hi);
}
static __device__ __forceinline__ int iclamp(int v, int lo, int hi) {
  return v < lo ? lo : (v > hi ? hi : v);
}

// ---------------- features fp32 [512][2500] -> f16 pixel-major [2500][512] ----------------
__global__ void k_cvt_feat(const float* __restrict__ src, _Float16* __restrict__ dst) {
  int i = blockIdx.x * blockDim.x + threadIdx.x;
  if (i >= CIN * NPIX) return;
  int c = i % CIN;
  int p = i / CIN;
  dst[i] = (_Float16)src[(size_t)c * NPIX + p];   // dst[p*512 + c]
}

// rpn_conv_w (O=512, I=512, 3, 3) fp32 -> wT f16 [tap][cin/2][cout][2] (K-pairs contiguous)
__global__ void k_prep_wconv(const float* __restrict__ w, _Float16* __restrict__ wT) {
  int i = blockIdx.x * blockDim.x + threadIdx.x;
  const int total = 9 * CIN * CIN;
  if (i >= total) return;
  int r = i & 1;
  int t = i >> 1;
  int o = t % CIN;   t /= CIN;
  int cp = t % (CIN / 2);
  int tap = t / (CIN / 2);
  int cin = cp * 2 + r;
  wT[i] = (_Float16)w[(size_t)o * (CIN * 9) + (size_t)cin * 9 + tap];
}

// ---------------- RPN 3x3 conv as implicit GEMM (WMMA f16) ----------------
// M = 2500 pixels, N = 512 outch, K = 9*512. One 16x16 C-tile per wave, 8 waves/block.
__global__ void k_rpn_conv(const _Float16* __restrict__ feat,   // [2500][512] f16
                           const _Float16* __restrict__ wT,     // [9][256][512][2] f16
                           const float* __restrict__ bias,      // [512]
                           _Float16* __restrict__ h)            // [2500][512] f16
{
  const int lane = threadIdx.x & 31;
  const int wave = threadIdx.x >> 5;
  const int half = lane >> 4;
  const int mrow = lane & 15;
  const int nl   = lane & 15;
  const int m0 = blockIdx.x * 16;
  const int n0 = blockIdx.y * 128 + wave * 16;
  const int m = m0 + mrow;
  const bool mok = (m < NPIX);
  const int py = mok ? (m / WFD) : 0;
  const int px = mok ? (m - py * WFD) : 0;
  const unsigned* featU = (const unsigned*)feat;   // [2500][256] dword pairs
  const unsigned* wTU   = (const unsigned*)wT;     // [9][256][512] dwords

  v8f acc = {};
  for (int tap = 0; tap < 9; ++tap) {
    const int kh = tap / 3, kw = tap - kh * 3;
    const int yy = py + kh - 1, xx = px + kw - 1;
    const bool valid = mok && (yy >= 0) && (yy < HFD) && (xx >= 0) && (xx < WFD);
    const int bsafe = valid ? (yy * WFD + xx) : 0;            // clamped, branch-free
    const unsigned* arow = featU + (size_t)bsafe * (CIN / 2);
    const unsigned* wtap = wTU + (size_t)tap * (CIN / 2) * CIN + (n0 + nl);
#pragma unroll 4
    for (int c0 = 0; c0 < CIN; c0 += 64) {
      Frag a0, b0, a1, b1;
#pragma unroll
      for (int j = 0; j < 8; ++j) {
        int kp = 8 * half + j;                                // A pair index within chunk
        unsigned av0 = arow[(c0 >> 1) + kp];
        unsigned av1 = arow[(c0 >> 1) + 16 + kp];
        a0.u[j] = valid ? av0 : 0u;
        a1.u[j] = valid ? av1 : 0u;
        int bp = 8 * half + j;                                // B pair index within chunk
        b0.u[j] = wtap[(size_t)((c0 >> 1) + bp) * CIN];
        b1.u[j] = wtap[(size_t)((c0 >> 1) + 16 + bp) * CIN];
      }
      acc = __builtin_amdgcn_wmma_f32_16x16x32_f16(false, a0.h, false, b0.h, (short)0, acc,
                                                   false, false);
      acc = __builtin_amdgcn_wmma_f32_16x16x32_f16(false, a1.h, false, b1.h, (short)0, acc,
                                                   false, false);
    }
  }
  const int n = n0 + nl;
  const float bn = bias[n];
#pragma unroll
  for (int v = 0; v < 8; ++v) {
    int mm = m0 + v + 8 * half;   // C/D layout: M = v + 8*half
    if (mm < NPIX) {
      float val = acc[v] + bn;
      h[(size_t)mm * CIN + n] = (_Float16)fmaxf(val, 0.f);
    }
  }
}

// ---------------- RPN heads (1x1 convs, tiny) + softmax + anchor decode ----------------
__global__ void k_rpn_heads(const _Float16* __restrict__ h,    // [2500][512]
                            const float* __restrict__ wcls,    // [18][512]
                            const float* __restrict__ bcls,    // [18]
                            const float* __restrict__ wreg,    // [36][512]
                            const float* __restrict__ breg,    // [36]
                            const float* __restrict__ anchors, // [22500][4]
                            float* __restrict__ scores,
                            float* __restrict__ rois)
{
  int idx = blockIdx.x * blockDim.x + threadIdx.x;
  if (idx >= NANCH) return;
  int a = idx % 9;
  int pix = idx / 9;
  const _Float16* hr = h + (size_t)pix * CIN;
  float l0 = bcls[a * 2 + 0], l1 = bcls[a * 2 + 1];
  float d0 = breg[a * 4 + 0], d1 = breg[a * 4 + 1];
  float d2 = breg[a * 4 + 2], d3 = breg[a * 4 + 3];
  const float* w0 = wcls + (size_t)(a * 2 + 0) * CIN;
  const float* w1 = wcls + (size_t)(a * 2 + 1) * CIN;
  const float* r0 = wreg + (size_t)(a * 4 + 0) * CIN;
  const float* r1 = wreg + (size_t)(a * 4 + 1) * CIN;
  const float* r2 = wreg + (size_t)(a * 4 + 2) * CIN;
  const float* r3 = wreg + (size_t)(a * 4 + 3) * CIN;
  for (int c = 0; c < CIN; ++c) {
    float hv = (float)hr[c];
    l0 += hv * w0[c]; l1 += hv * w1[c];
    d0 += hv * r0[c]; d1 += hv * r1[c];
    d2 += hv * r2[c]; d3 += hv * r3[c];
  }
  float score = 1.f / (1.f + __expf(l0 - l1));   // softmax(...)[1]
  const float* an = anchors + (size_t)idx * 4;
  float ah = an[2] - an[0], aw = an[3] - an[1];
  float acy = an[0] + 0.5f * ah, acx = an[1] + 0.5f * aw;
  float dh = clampf(d2, -BBOX_CLIP, BBOX_CLIP);
  float dw = clampf(d3, -BBOX_CLIP, BBOX_CLIP);
  float pcy = acy + d0 * ah, pcx = acx + d1 * aw;
  float ph = ah * __expf(dh), pw = aw * __expf(dw);
  rois[idx * 4 + 0] = clampf(pcy - 0.5f * ph, 0.f, 800.f);
  rois[idx * 4 + 1] = clampf(pcx - 0.5f * pw, 0.f, 800.f);
  rois[idx * 4 + 2] = clampf(pcy + 0.5f * ph, 0.f, 800.f);
  rois[idx * 4 + 3] = clampf(pcx + 0.5f * pw, 0.f, 800.f);
  scores[idx] = score;
}

// ---------------- sequential NMS, one workgroup ----------------
__global__ void __launch_bounds__(1024)
k_nms(const float* __restrict__ rois, const float* __restrict__ scores,
      float* __restrict__ sbuf, float* __restrict__ rois128,
      float* __restrict__ ok128, float* __restrict__ out_rois)
{
  const int tid = threadIdx.x;
  __shared__ float rmax[1024];
  __shared__ int   ridx[1024];
  __shared__ float pbox[4];
  for (int j = tid; j < NANCH; j += 1024) sbuf[j] = scores[j];
  __syncthreads();
  for (int it = 0; it < 300; ++it) {
    float bm = -1e30f; int bi = 0;
    for (int j = tid; j < NANCH; j += 1024) {
      float v = sbuf[j];
      if (v > bm) { bm = v; bi = j; }
    }
    rmax[tid] = bm; ridx[tid] = bi;
    __syncthreads();
    for (int s = 512; s > 0; s >>= 1) {
      if (tid < s) {
        if (rmax[tid + s] > rmax[tid]) { rmax[tid] = rmax[tid + s]; ridx[tid] = ridx[tid + s]; }
      }
      __syncthreads();
    }
    if (tid == 0) {
      int i = ridx[0];
      float ok = (rmax[0] > -1e29f) ? 1.f : 0.f;
      pbox[0] = rois[i * 4 + 0]; pbox[1] = rois[i * 4 + 1];
      pbox[2] = rois[i * 4 + 2]; pbox[3] = rois[i * 4 + 3];
      if (it < NROI) {
        for (int j2 = 0; j2 < 4; ++j2) {
          float bv = pbox[j2] * ok;
          rois128[it * 4 + j2] = bv;
          out_rois[it * 4 + j2] = bv;
        }
        ok128[it] = ok;
      }
    }
    __syncthreads();
    float y1 = pbox[0], x1 = pbox[1], y2 = pbox[2], x2 = pbox[3];
    float pa = (y2 - y1) * (x2 - x1);
    for (int j = tid; j < NANCH; j += 1024) {
      float by1 = rois[j * 4 + 0], bx1 = rois[j * 4 + 1];
      float by2 = rois[j * 4 + 2], bx2 = rois[j * 4 + 3];
      float yy1 = fmaxf(y1, by1), xx1 = fmaxf(x1, bx1);
      float yy2 = fminf(y2, by2), xx2 = fminf(x2, bx2);
      float inter = fmaxf(yy2 - yy1, 0.f) * fmaxf(xx2 - xx1, 0.f);
      float ba = (by2 - by1) * (bx2 - bx1);
      float iou = inter / (pa + ba - inter + 1e-9f);
      if (iou > 0.7f) sbuf[j] = -1e30f;
    }
    __syncthreads();
  }
}

// ---------------- ROI align 7x7 (2x2 bilinear samples) -> f16 GEMM input ----------------
__global__ void k_roi_align(const float* __restrict__ feat,     // [512][50][50] fp32
                            const float* __restrict__ rois128,  // image coords
                            _Float16* __restrict__ X)           // [128][25088] f16
{
  int idx = blockIdx.x * blockDim.x + threadIdx.x;
  const int total = NROI * CIN * 49;
  if (idx >= total) return;
  int r = idx / (CIN * 49);
  int rem = idx - r * (CIN * 49);
  int c = rem / 49;
  int p = rem - c * 49;
  int phh = p / 7, pww = p - phh * 7;
  const float* bx = rois128 + r * 4;
  float y1 = bx[0] * 0.0625f, x1 = bx[1] * 0.0625f;
  float y2 = bx[2] * 0.0625f, x2 = bx[3] * 0.0625f;
  float bh = (y2 - y1) / 7.f, bw = (x2 - x1) / 7.f;
  const float* fc = feat + (size_t)c * NPIX;
  float acc = 0.f;
#pragma unroll
  for (int sy = 0; sy < 2; ++sy) {
#pragma unroll
    for (int sx = 0; sx < 2; ++sx) {
      float y = y1 + bh * ((float)phh + (sy + 0.5f) * 0.5f);
      float x = x1 + bw * ((float)pww + (sx + 0.5f) * 0.5f);
      float yf = floorf(y), xf = floorf(x);
      float ly = y - yf, lx = x - xf;
      int y0 = iclamp((int)yf, 0, HFD - 1);
      int y1i = iclamp((int)yf + 1, 0, HFD - 1);
      int x0 = iclamp((int)xf, 0, WFD - 1);
      int x1i = iclamp((int)xf + 1, 0, WFD - 1);
      float f00 = fc[y0 * WFD + x0], f01 = fc[y0 * WFD + x1i];
      float f10 = fc[y1i * WFD + x0], f11 = fc[y1i * WFD + x1i];
      acc += f00 * (1.f - ly) * (1.f - lx) + f01 * (1.f - ly) * lx +
             f10 * ly * (1.f - lx) + f11 * ly * lx;
    }
  }
  X[(size_t)r * FC1K + c * 49 + p] = (_Float16)(acc * 0.25f);
}

// ---------------- FC GEMM: C[M=128,N] = act(A[128,K]f16 @ B[K,N]f32 + bias) ----------------
// 8 waves x one 16-row M-tile each; one 16-wide N tile per block; 2 WMMA per K-iter.
__global__ void k_fc_gemm(const _Float16* __restrict__ A, const float* __restrict__ B,
                          const float* __restrict__ bias, _Float16* __restrict__ C,
                          int N, int K, int relu)
{
  const int lane = threadIdx.x & 31;
  const int wave = threadIdx.x >> 5;
  const int half = lane >> 4;
  const int m = wave * 16 + (lane & 15);          // A row for this lane
  const int ncol = blockIdx.x * 16 + (lane & 15);
  const float* __restrict__ Bc = B + ncol;
  const unsigned* __restrict__ Ar = (const unsigned*)(A + (size_t)m * K);
  v8f acc = {};
  for (int k0 = 0; k0 < K; k0 += 64) {
    if (k0 + 128 < K)
      __builtin_prefetch((const void*)(Bc + (size_t)(k0 + 128) * N), 0, 0);
    Frag a0, b0, a1, b1;
#pragma unroll
    for (int j = 0; j < 8; ++j) {
      int kp = 8 * half + j;                                 // A pair index
      a0.u[j] = Ar[(k0 >> 1) + kp];
      a1.u[j] = Ar[(k0 >> 1) + 16 + kp];
      int kk = 16 * half + 2 * j;                            // B row index
      b0.h[2 * j]     = (_Float16)Bc[(size_t)(k0 + kk) * N];
      b0.h[2 * j + 1] = (_Float16)Bc[(size_t)(k0 + kk + 1) * N];
      b1.h[2 * j]     = (_Float16)Bc[(size_t)(k0 + 32 + kk) * N];
      b1.h[2 * j + 1] = (_Float16)Bc[(size_t)(k0 + 32 + kk + 1) * N];
    }
    acc = __builtin_amdgcn_wmma_f32_16x16x32_f16(false, a0.h, false, b0.h, (short)0, acc,
                                                 false, false);
    acc = __builtin_amdgcn_wmma_f32_16x16x32_f16(false, a1.h, false, b1.h, (short)0, acc,
                                                 false, false);
  }
  const float bn = bias[ncol];
#pragma unroll
  for (int v = 0; v < 8; ++v) {
    int mm = wave * 16 + v + 8 * half;
    float val = acc[v] + bn;
    if (relu) val = fmaxf(val, 0.f);
    C[(size_t)mm * N + ncol] = (_Float16)val;
  }
}

// ---------------- final heads: cls logits (21) and box deltas (84) ----------------
__global__ void k_heads(const _Float16* __restrict__ Y2,
                        const float* __restrict__ Wc, const float* __restrict__ bc,
                        const float* __restrict__ Wb, const float* __restrict__ bb,
                        float* __restrict__ clsl, float* __restrict__ rdel)
{
  int idx = blockIdx.x * blockDim.x + threadIdx.x;
  if (idx >= NROI * 105) return;
  int r = idx / 105;
  int o = idx - r * 105;
  const _Float16* yr = Y2 + (size_t)r * FCN;
  if (o < NCLS) {
    float s = bc[o];
    for (int k = 0; k < FCN; ++k) s += (float)yr[k] * Wc[(size_t)k * NCLS + o];
    clsl[r * NCLS + o] = s;
  } else {
    int ob = o - NCLS;
    float s = bb[ob];
    for (int k = 0; k < FCN; ++k) s += (float)yr[k] * Wb[(size_t)k * (NCLS * 4) + ob];
    rdel[r * (NCLS * 4) + ob] = s;
  }
}

// ---------------- final decode / softmax / keep mask -> d_out ----------------
__global__ void k_final(const float* __restrict__ clsl, const float* __restrict__ rdel,
                        const float* __restrict__ rois128, const float* __restrict__ ok128,
                        float* __restrict__ out)
{
  int idx = blockIdx.x * blockDim.x + threadIdx.x;
  if (idx >= NROI * (NCLS - 1)) return;
  int r = idx / (NCLS - 1);
  int c = idx - r * (NCLS - 1) + 1;
  const float* lg = clsl + r * NCLS;
  float mx = lg[0];
  for (int k = 1; k < NCLS; ++k) mx = fmaxf(mx, lg[k]);
  float se = 0.f;
  for (int k = 0; k < NCLS; ++k) se += __expf(lg[k] - mx);
  float prob = __expf(lg[c] - mx) / se;
  const float* bx = rois128 + r * 4;
  float h0 = bx[2] - bx[0], w0 = bx[3] - bx[1];
  float cy = bx[0] + 0.5f * h0, cx = bx[1] + 0.5f * w0;
  const float* d = rdel + r * 84 + c * 4;
  float dy = d[0] * 0.1f, dx = d[1] * 0.1f;
  float dh = clampf(d[2] * 0.2f, -BBOX_CLIP, BBOX_CLIP);
  float dw = clampf(d[3] * 0.2f, -BBOX_CLIP, BBOX_CLIP);
  float pcy = cy + dy * h0, pcx = cx + dx * w0;
  float phh = h0 * __expf(dh), pww = w0 * __expf(dw);
  float by1 = pcy - 0.5f * phh, bx1 = pcx - 0.5f * pww;
  float by2 = pcy + 0.5f * phh, bx2 = pcx + 0.5f * pww;
  float hh = by2 - by1, ww = bx2 - bx1;
  float keep = (prob > 0.05f && hh >= 1.f && ww >= 1.f && ok128[r] > 0.5f) ? 1.f : 0.f;
  out[idx] = prob;                            // scores_out [0,2560)
  float* ob = out + 2560;                     // boxes [2560,12800)
  ob[idx * 4 + 0] = by1; ob[idx * 4 + 1] = bx1;
  ob[idx * 4 + 2] = by2; ob[idx * 4 + 3] = bx2;
  out[12800 + idx] = (float)(c - 1);          // cls_idxes
  out[15360 + idx] = keep;                    // keep
}

extern "C" void kernel_launch(void* const* d_in, const int* in_sizes, int n_in,
                              void* d_out, int out_size, void* d_ws, size_t ws_size,
                              hipStream_t stream) {
  (void)in_sizes; (void)n_in; (void)out_size; (void)ws_size;
  const float* features   = (const float*)d_in[0];
  const float* anchors    = (const float*)d_in[1];
  const float* rpn_conv_w = (const float*)d_in[2];
  const float* rpn_conv_b = (const float*)d_in[3];
  const float* rpn_cls_w  = (const float*)d_in[4];
  const float* rpn_cls_b  = (const float*)d_in[5];
  const float* rpn_reg_w  = (const float*)d_in[6];
  const float* rpn_reg_b  = (const float*)d_in[7];
  const float* W1 = (const float*)d_in[8];
  const float* b1 = (const float*)d_in[9];
  const float* W2 = (const float*)d_in[10];
  const float* b2 = (const float*)d_in[11];
  const float* Wc = (const float*)d_in[12];
  const float* bc = (const float*)d_in[13];
  const float* Wb = (const float*)d_in[14];
  const float* bb = (const float*)d_in[15];
  float* out = (float*)d_out;

  char* ws = (char*)d_ws;
  size_t off = 0;
  auto alloc = [&](size_t bytes) -> void* {
    void* p = ws + off;
    off = (off + bytes + 255) & ~(size_t)255;
    return p;
  };
  _Float16* feat16 = (_Float16*)alloc((size_t)CIN * NPIX * 2);     // pixel-major
  _Float16* wconvT = (_Float16*)alloc((size_t)9 * CIN * CIN * 2);  // pair-packed
  _Float16* h16    = (_Float16*)alloc((size_t)NPIX * CIN * 2);
  float* scores    = (float*)alloc((size_t)NANCH * 4);
  float* sbuf      = (float*)alloc((size_t)NANCH * 4);
  float* rois      = (float*)alloc((size_t)NANCH * 16);
  float* rois128   = (float*)alloc((size_t)NROI * 16);
  float* ok128     = (float*)alloc((size_t)NROI * 4);
  _Float16* X16    = (_Float16*)alloc((size_t)NROI * FC1K * 2);
  _Float16* Y1     = (_Float16*)alloc((size_t)NROI * FCN * 2);
  _Float16* Y2     = (_Float16*)alloc((size_t)NROI * FCN * 2);
  float* clsl      = (float*)alloc((size_t)NROI * NCLS * 4);
  float* rdel      = (float*)alloc((size_t)NROI * NCLS * 16);

  k_cvt_feat<<<(CIN * NPIX + 255) / 256, 256, 0, stream>>>(features, feat16);
  k_prep_wconv<<<(9 * CIN * CIN + 255) / 256, 256, 0, stream>>>(rpn_conv_w, wconvT);
  k_rpn_conv<<<dim3(157, 4), 256, 0, stream>>>(feat16, wconvT, rpn_conv_b, h16);
  k_rpn_heads<<<(NANCH + 255) / 256, 256, 0, stream>>>(h16, rpn_cls_w, rpn_cls_b,
                                                       rpn_reg_w, rpn_reg_b, anchors,
                                                       scores, rois);
  k_nms<<<1, 1024, 0, stream>>>(rois, scores, sbuf, rois128, ok128, out + 17920);
  k_roi_align<<<(NROI * CIN * 49 + 255) / 256, 256, 0, stream>>>(features, rois128, X16);
  k_fc_gemm<<<FCN / 16, 256, 0, stream>>>(X16, W1, b1, Y1, FCN, FC1K, 1);
  k_fc_gemm<<<FCN / 16, 256, 0, stream>>>(Y1, W2, b2, Y2, FCN, FCN, 1);
  k_heads<<<(NROI * 105 + 255) / 256, 256, 0, stream>>>(Y2, Wc, bc, Wb, bb, clsl, rdel);
  k_final<<<(NROI * 20 + 255) / 256, 256, 0, stream>>>(clsl, rdel, rois128, ok128, out);
}
